// DeepSeekV3MoE_807453852024
// MI455X (gfx1250) — matrix operand order repaired and checked by reading, not compiled, not run on hardware
//
#include <hip/hip_runtime.h>
#include <hip/hip_bf16.h>

// Problem constants (from the reference).
#define HDIM 2048
#define IDIM 1408
#define NEXP 16
#define SIDIM 2816
#define NTOK 4096
#define NTOPK 2
#define MAXSLOTS (NTOK * NTOPK)   // 8192
#define TILEM 128                 // rows per M-tile (256-thread block, 8 waves)
#define MAXTILES 96               // >= ceil(8192/128) + 15 padding tiles
#define LDSROW 40                 // padded LDS row stride in halves (80B, 16B-aligned)

typedef __attribute__((ext_vector_type(16))) __bf16 v16bf;
typedef __attribute__((ext_vector_type(2)))  __bf16 bf16x2;
typedef __attribute__((ext_vector_type(8)))  float  v8f;

union Frag16 { v16bf v; uint4 q[2]; unsigned short h[16]; };

__device__ __forceinline__ unsigned short f2bf(float f) {
  return __builtin_bit_cast(unsigned short, (__bf16)f);
}
__device__ __forceinline__ unsigned int f2bf2(float a, float b) {
  bf16x2 p;
  p[0] = (__bf16)a;
  p[1] = (__bf16)b;
  return __builtin_bit_cast(unsigned int, p);
}

// 16-bit A fragment (16x32): lane l holds M=l&15; halves 0-7 -> K=kb..kb+7,
// halves 8-15 -> K=kb+16..kb+23, kb = 8*(l>>4).  Two ds_load_b128.
__device__ __forceinline__ v16bf frag_a(const unsigned short* row, int lane) {
  Frag16 f;
  const int kb = 8 * (lane >> 4);
  f.q[0] = *(const uint4*)(row + kb);
  f.q[1] = *(const uint4*)(row + kb + 16);
  return f.v;
}

// 16-bit B fragment (32x16): lane l holds N=l&15; halves 0-7 -> K=kb..kb+7,
// halves 8-15 -> K=kb+16..kb+23, kb = 8*(l>>4).  B staged transposed in LDS as [n][k] -> two ds_load_b128.
__device__ __forceinline__ v16bf frag_b(const unsigned short* row, int lane) {
  Frag16 f;
  const int kb = 8 * (lane >> 4);
  f.q[0] = *(const uint4*)(row + kb);
  f.q[1] = *(const uint4*)(row + kb + 16);
  return f.v;
}

// Async copy of 32 contiguous bytes (bf16 row chunk) global -> LDS.
// INST_OFFSET applies to both global and LDS addresses.
__device__ __forceinline__ void async_copy32(unsigned ldsAddr,
                                             unsigned long long gAddr) {
  asm volatile(
      "global_load_async_to_lds_b128 %0, %1, off\n\t"
      "global_load_async_to_lds_b128 %0, %1, off offset:16"
      :: "v"(ldsAddr), "v"(gAddr) : "memory");
}
__device__ __forceinline__ void async_fence() {
  asm volatile("s_wait_asynccnt 0x0" ::: "memory");
}

// ---------------------------------------------------------------- router ----
__global__ __launch_bounds__(128)
void router_kernel(const float* __restrict__ x, const float* __restrict__ rw,
                   const float* __restrict__ rb, int* __restrict__ topkId,
                   float* __restrict__ topkW, int* __restrict__ counts) {
  const int t = blockIdx.x * 4 + (threadIdx.x >> 5);
  const int lane = threadIdx.x & 31;
  float acc[NEXP];
#pragma unroll
  for (int e = 0; e < NEXP; ++e) acc[e] = 0.f;
  const float* xr = x + (size_t)t * HDIM;
  for (int h = lane; h < HDIM; h += 32) {
    const float xv = xr[h];
#pragma unroll
    for (int e = 0; e < NEXP; ++e) acc[e] += xv * rw[e * HDIM + h];
  }
#pragma unroll
  for (int e = 0; e < NEXP; ++e)
#pragma unroll
    for (int off = 16; off > 0; off >>= 1)
      acc[e] += __shfl_xor(acc[e], off, 32);
  if (lane == 0) {
    float sc[NEXP];
#pragma unroll
    for (int e = 0; e < NEXP; ++e) {
      const float l = acc[e] + rb[e];
      sc[e] = 1.f / (1.f + __expf(-l));
    }
    int i0 = 0; float b0 = sc[0];
    for (int e = 1; e < NEXP; ++e) if (sc[e] > b0) { b0 = sc[e]; i0 = e; }
    int i1 = (i0 == 0) ? 1 : 0; float b1 = sc[i1];
    for (int e = 0; e < NEXP; ++e)
      if (e != i0 && sc[e] > b1) { b1 = sc[e]; i1 = e; }
    const float inv = 1.f / (b0 + b1 + 1e-8f);
    topkId[t * 2 + 0] = i0;  topkW[t * 2 + 0] = b0 * inv;
    topkId[t * 2 + 1] = i1;  topkW[t * 2 + 1] = b1 * inv;
    atomicAdd(&counts[i0], 1);
    atomicAdd(&counts[i1], 1);
  }
}

__global__ void zero_counts_kernel(int* counts) {
  if (threadIdx.x < NEXP) counts[threadIdx.x] = 0;
}

// Single-thread planning: prefix sum + 128-row tile map per expert.
__global__ void plan_kernel(const int* __restrict__ counts, int* __restrict__ offsets,
                            int* __restrict__ cursor, int* __restrict__ tileExpert,
                            int* __restrict__ tileSlotBase, int* __restrict__ tileRows) {
  if (threadIdx.x != 0 || blockIdx.x != 0) return;
  int off = 0, nt = 0;
  for (int e = 0; e < NEXP; ++e) {
    offsets[e] = off;
    cursor[e] = off;
    const int c = counts[e];
    for (int r = 0; r < c && nt < MAXTILES; r += TILEM) {
      tileExpert[nt] = e;
      tileSlotBase[nt] = off + r;
      tileRows[nt] = (c - r < TILEM) ? (c - r) : TILEM;
      ++nt;
    }
    off += c;
  }
  offsets[NEXP] = off;
  for (int i = nt; i < MAXTILES; ++i) tileRows[i] = 0;
}

__global__ void scatter_kernel(const int* __restrict__ topkId,
                               const float* __restrict__ topkW,
                               int* __restrict__ cursor, int* __restrict__ slotToken,
                               float* __restrict__ slotW) {
  const int t = blockIdx.x * 256 + threadIdx.x;
  if (t >= NTOK) return;
#pragma unroll
  for (int k = 0; k < NTOPK; ++k) {
    const int e = topkId[t * 2 + k];
    const int slot = atomicAdd(&cursor[e], 1);
    slotToken[slot] = t;
    slotW[slot] = topkW[t * 2 + k];
  }
}

__global__ __launch_bounds__(256)
void xcvt_kernel(const float* __restrict__ x, unsigned short* __restrict__ xb) {
  const size_t i = ((size_t)blockIdx.x * 256 + threadIdx.x) * 8;
  const float4 a = ((const float4*)(x + i))[0];
  const float4 b = ((const float4*)(x + i))[1];
  unsigned int r[4] = {f2bf2(a.x, a.y), f2bf2(a.z, a.w),
                       f2bf2(b.x, b.y), f2bf2(b.z, b.w)};
  *(uint4*)(xb + i) = *(const uint4*)r;
}

// ------------------------------------------- fused SwiGLU gate/up (WMMA) ----
// Block = 256 threads (8 waves). Block tile: 128(M) x 32(N), K-step 32.
// Double-buffered LDS pipeline: async A copy + raw global B loads for tile
// k+1 are issued before computing tile k; B cvt+store lands after the WMMAs.
template <bool ROUTED>
__global__ __launch_bounds__(256)
void gateup_kernel(const unsigned short* __restrict__ xt,   // [NTOK,H] bf16
                   const float* __restrict__ Wg0,           // [E,H,I] or [H,SI]
                   const float* __restrict__ Wu0,
                   unsigned short* __restrict__ hdn,        // [rows, ldN] bf16
                   const int* __restrict__ slotToken,
                   const int* __restrict__ tileExpert,
                   const int* __restrict__ tileSlotBase,
                   const int* __restrict__ tileRows, int ldN) {
  __shared__ __align__(16) unsigned short lA[2][TILEM * LDSROW];
  __shared__ __align__(16) unsigned short lBg[2][32 * LDSROW];
  __shared__ __align__(16) unsigned short lBu[2][32 * LDSROW];

  const int tile = blockIdx.x;
  int rows, slotBase;
  const float *Wg, *Wu;
  if (ROUTED) {
    rows = tileRows[tile];
    if (rows == 0) return;
    slotBase = tileSlotBase[tile];
    const int e = tileExpert[tile];
    Wg = Wg0 + (size_t)e * HDIM * IDIM;
    Wu = Wu0 + (size_t)e * HDIM * IDIM;
  } else {
    rows = TILEM; slotBase = tile * TILEM; Wg = Wg0; Wu = Wu0;
  }

  const int n0 = blockIdx.y * 32;
  const int tid = threadIdx.x;
  const int lane = tid & 31;
  const int wave = tid >> 5;

  // A staging: thread -> (row 0..127, 16-half chunk).
  const int ar = tid >> 1;
  const int ac = (tid & 1) * 16;
  const int arow = (ar < rows) ? ar : (rows - 1);
  const int token = ROUTED ? slotToken[slotBase + arow] : (slotBase + arow);
  const unsigned ldsA0 = (unsigned)(uintptr_t)(&lA[0][ar * LDSROW + ac]);
  const unsigned ldsA1 = (unsigned)(uintptr_t)(&lA[1][ar * LDSROW + ac]);
  const unsigned short* gA = xt + (size_t)token * HDIM + ac;

  // B staging: thread -> (even k 0..30, even n 0..30), 2x2 micro-block.
  const int bk = ((tid >> 4) & 15) * 2;
  const int bn = (tid & 15) * 2;

  auto loadB = [&](int kk, float2& g0, float2& g1, float2& u0, float2& u1) {
    const float* gp = Wg + (size_t)(kk + bk) * ldN + n0 + bn;
    const float* up = Wu + (size_t)(kk + bk) * ldN + n0 + bn;
    g0 = *(const float2*)gp;
    g1 = *(const float2*)(gp + ldN);
    u0 = *(const float2*)up;
    u1 = *(const float2*)(up + ldN);
  };
  auto storeB = [&](int buf, float2 g0, float2 g1, float2 u0, float2 u1) {
    *(unsigned int*)(&lBg[buf][(bn + 0) * LDSROW + bk]) = f2bf2(g0.x, g1.x);
    *(unsigned int*)(&lBg[buf][(bn + 1) * LDSROW + bk]) = f2bf2(g0.y, g1.y);
    *(unsigned int*)(&lBu[buf][(bn + 0) * LDSROW + bk]) = f2bf2(u0.x, u1.x);
    *(unsigned int*)(&lBu[buf][(bn + 1) * LDSROW + bk]) = f2bf2(u0.y, u1.y);
  };

  v8f cg0 = {}, cg1 = {}, cu0 = {}, cu1 = {};
  float2 g0, g1, u0, u1;

  // Prologue: stage buffer 0 for kk = 0.
  async_copy32(ldsA0, (unsigned long long)(uintptr_t)gA);
  loadB(0, g0, g1, u0, u1);
  storeB(0, g0, g1, u0, u1);

  for (int kk = 0; kk < HDIM; kk += 32) {
    const int buf = (kk >> 5) & 1;
    async_fence();
    __syncthreads();  // buffer `buf` staged; everyone done reading `buf^1`

    const bool nxt = (kk + 32) < HDIM;
    if (nxt) {
      async_copy32(buf ? ldsA0 : ldsA1,
                   (unsigned long long)(uintptr_t)(gA + kk + 32));
      loadB(kk + 32, g0, g1, u0, u1);  // global loads in flight during WMMA
    }

    const v16bf a   = frag_a(&lA[buf][(wave * 16 + (lane & 15)) * LDSROW], lane);
    const v16bf bg0 = frag_b(&lBg[buf][((lane & 15)) * LDSROW], lane);
    const v16bf bg1 = frag_b(&lBg[buf][(16 + (lane & 15)) * LDSROW], lane);
    const v16bf bu0 = frag_b(&lBu[buf][((lane & 15)) * LDSROW], lane);
    const v16bf bu1 = frag_b(&lBu[buf][(16 + (lane & 15)) * LDSROW], lane);

    cg0 = __builtin_amdgcn_wmma_f32_16x16x32_bf16(false, a, false, bg0, (short)0, cg0, false, false);
    cg1 = __builtin_amdgcn_wmma_f32_16x16x32_bf16(false, a, false, bg1, (short)0, cg1, false, false);
    cu0 = __builtin_amdgcn_wmma_f32_16x16x32_bf16(false, a, false, bu0, (short)0, cu0, false, false);
    cu1 = __builtin_amdgcn_wmma_f32_16x16x32_bf16(false, a, false, bu1, (short)0, cu1, false, false);

    if (nxt) storeB(buf ^ 1, g0, g1, u0, u1);
  }

  // Epilogue: silu(gate)*up -> bf16.  C layout: VGPR v, lane l -> M=v+8*(l>>4), N=l&15.
  const int mBase = wave * 16 + 8 * (lane >> 4);
  const int nl = lane & 15;
#pragma unroll
  for (int v = 0; v < 8; ++v) {
    const int m = mBase + v;
    if (m < rows) {
      const size_t orow = (size_t)(slotBase + m) * ldN;
      const float g0v = cg0[v];
      const float s0 = g0v / (1.f + __expf(-g0v));
      hdn[orow + n0 + nl] = f2bf(s0 * cu0[v]);
      const float g1v = cg1[v];
      const float s1 = g1v / (1.f + __expf(-g1v));
      hdn[orow + n0 + 16 + nl] = f2bf(s1 * cu1[v]);
    }
  }
}

// ------------------------------------------------- down projection (WMMA) ---
// Shared: plain stores (first full write of out). Routed: weighted f32 atomics.
template <bool ROUTED>
__global__ __launch_bounds__(256)
void down_kernel(const unsigned short* __restrict__ hdn,  // [rows, K] bf16
                 const float* __restrict__ Wd0,           // [E,I,H] or [SI,H]
                 float* __restrict__ out,                 // [NTOK, H]
                 const int* __restrict__ slotToken, const float* __restrict__ slotW,
                 const int* __restrict__ tileExpert,
                 const int* __restrict__ tileSlotBase,
                 const int* __restrict__ tileRows, int K) {
  __shared__ __align__(16) unsigned short lA[2][TILEM * LDSROW];
  __shared__ __align__(16) unsigned short lB[2][32 * LDSROW];

  const int tile = blockIdx.x;
  int rows, slotBase;
  const float* Wd;
  if (ROUTED) {
    rows = tileRows[tile];
    if (rows == 0) return;
    slotBase = tileSlotBase[tile];
    Wd = Wd0 + (size_t)tileExpert[tile] * IDIM * HDIM;
  } else {
    rows = TILEM; slotBase = tile * TILEM; Wd = Wd0;
  }

  const int n0 = blockIdx.y * 32;
  const int tid = threadIdx.x;
  const int lane = tid & 31;
  const int wave = tid >> 5;

  const int ar = tid >> 1;
  const int ac = (tid & 1) * 16;
  const int arow = slotBase + ((ar < rows) ? ar : (rows - 1));
  const unsigned ldsA0 = (unsigned)(uintptr_t)(&lA[0][ar * LDSROW + ac]);
  const unsigned ldsA1 = (unsigned)(uintptr_t)(&lA[1][ar * LDSROW + ac]);
  const unsigned short* gA = hdn + (size_t)arow * K + ac;

  const int bk = ((tid >> 4) & 15) * 2;
  const int bn = (tid & 15) * 2;

  auto loadB = [&](int kk, float2& w0, float2& w1) {
    const float* wp = Wd + (size_t)(kk + bk) * HDIM + n0 + bn;
    w0 = *(const float2*)wp;
    w1 = *(const float2*)(wp + HDIM);
  };
  auto storeB = [&](int buf, float2 w0, float2 w1) {
    *(unsigned int*)(&lB[buf][(bn + 0) * LDSROW + bk]) = f2bf2(w0.x, w1.x);
    *(unsigned int*)(&lB[buf][(bn + 1) * LDSROW + bk]) = f2bf2(w0.y, w1.y);
  };

  v8f c0 = {}, c1 = {};
  float2 w0, w1;

  async_copy32(ldsA0, (unsigned long long)(uintptr_t)gA);
  loadB(0, w0, w1);
  storeB(0, w0, w1);

  for (int kk = 0; kk < K; kk += 32) {
    const int buf = (kk >> 5) & 1;
    async_fence();
    __syncthreads();

    const bool nxt = (kk + 32) < K;
    if (nxt) {
      async_copy32(buf ? ldsA0 : ldsA1,
                   (unsigned long long)(uintptr_t)(gA + kk + 32));
      loadB(kk + 32, w0, w1);
    }

    const v16bf a  = frag_a(&lA[buf][(wave * 16 + (lane & 15)) * LDSROW], lane);
    const v16bf b0 = frag_b(&lB[buf][((lane & 15)) * LDSROW], lane);
    const v16bf b1 = frag_b(&lB[buf][(16 + (lane & 15)) * LDSROW], lane);

    c0 = __builtin_amdgcn_wmma_f32_16x16x32_bf16(false, a, false, b0, (short)0, c0, false, false);
    c1 = __builtin_amdgcn_wmma_f32_16x16x32_bf16(false, a, false, b1, (short)0, c1, false, false);

    if (nxt) storeB(buf ^ 1, w0, w1);
  }

  const int mBase = wave * 16 + 8 * (lane >> 4);
  const int nl = lane & 15;
#pragma unroll
  for (int v = 0; v < 8; ++v) {
    const int m = mBase + v;
    if (m < rows) {
      if (ROUTED) {
        const int slot = slotBase + m;
        const int token = slotToken[slot];
        const float w = slotW[slot];
        atomicAdd(&out[(size_t)token * HDIM + n0 + nl], w * c0[v]);
        atomicAdd(&out[(size_t)token * HDIM + n0 + 16 + nl], w * c1[v]);
      } else {
        out[(size_t)(slotBase + m) * HDIM + n0 + nl] = c0[v];
        out[(size_t)(slotBase + m) * HDIM + n0 + 16 + nl] = c1[v];
      }
    }
  }
}

// ----------------------------------------------------------------- launch ---
extern "C" void kernel_launch(void* const* d_in, const int* in_sizes, int n_in,
                              void* d_out, int out_size, void* d_ws, size_t ws_size,
                              hipStream_t stream) {
  const float* x   = (const float*)d_in[0];
  const float* rw  = (const float*)d_in[1];
  const float* rb  = (const float*)d_in[2];
  const float* wg  = (const float*)d_in[3];
  const float* wu  = (const float*)d_in[4];
  const float* wd  = (const float*)d_in[5];
  const float* sg  = (const float*)d_in[6];
  const float* su  = (const float*)d_in[7];
  const float* sd  = (const float*)d_in[8];
  float* out = (float*)d_out;

  char* ws = (char*)d_ws;
  unsigned short* xb   = (unsigned short*)ws;  ws += (size_t)NTOK * HDIM * 2;
  unsigned short* hdnS = (unsigned short*)ws;  ws += (size_t)NTOK * SIDIM * 2;
  unsigned short* hdnR = (unsigned short*)ws;  ws += (size_t)MAXSLOTS * IDIM * 2;
  int*   slotToken = (int*)ws;    ws += MAXSLOTS * 4;
  float* slotW     = (float*)ws;  ws += MAXSLOTS * 4;
  int*   topkId    = (int*)ws;    ws += NTOK * 2 * 4;
  float* topkW     = (float*)ws;  ws += NTOK * 2 * 4;
  int*   counts    = (int*)ws;    ws += 64;
  int*   offsets   = (int*)ws;    ws += 128;
  int*   cursor    = (int*)ws;    ws += 64;
  int*   tileExpert   = (int*)ws; ws += MAXTILES * 4;
  int*   tileSlotBase = (int*)ws; ws += MAXTILES * 4;
  int*   tileRows     = (int*)ws; ws += MAXTILES * 4;

  zero_counts_kernel<<<1, 32, 0, stream>>>(counts);
  router_kernel<<<NTOK / 4, 128, 0, stream>>>(x, rw, rb, topkId, topkW, counts);
  plan_kernel<<<1, 1, 0, stream>>>(counts, offsets, cursor, tileExpert,
                                   tileSlotBase, tileRows);
  scatter_kernel<<<NTOK / 256, 256, 0, stream>>>(topkId, topkW, cursor,
                                                 slotToken, slotW);
  xcvt_kernel<<<(NTOK * HDIM / 8) / 256, 256, 0, stream>>>(x, xb);

  dim3 gS(NTOK / TILEM, SIDIM / 32);
  gateup_kernel<false><<<gS, 256, 0, stream>>>(xb, sg, su, hdnS, nullptr,
                                               nullptr, nullptr, nullptr, SIDIM);
  dim3 gR(MAXTILES, IDIM / 32);
  gateup_kernel<true><<<gR, 256, 0, stream>>>(xb, wg, wu, hdnR, slotToken,
                                              tileExpert, tileSlotBase, tileRows,
                                              IDIM);
  // Shared down-proj writes every output element (plain stores), then routed
  // down-proj accumulates weighted results with f32 atomics (stream-ordered).
  dim3 dS(NTOK / TILEM, HDIM / 32);
  down_kernel<false><<<dS, 256, 0, stream>>>(hdnS, sd, out, nullptr, nullptr,
                                             nullptr, nullptr, nullptr, SIDIM);
  dim3 dR(MAXTILES, HDIM / 32);
  down_kernel<true><<<dR, 256, 0, stream>>>(hdnR, wd, out, slotToken, slotW,
                                            tileExpert, tileSlotBase, tileRows,
                                            IDIM);
}
